// VectorQuantizerLight_14766097563862
// MI455X (gfx1250) — compile-verified, matrix operand order checked by reading, not run on hardware
//
#include <hip/hip_runtime.h>
#include <hip/hip_bf16.h>
#include <math.h>

typedef __attribute__((ext_vector_type(16))) _Float16 v16h;
typedef __attribute__((ext_vector_type(8)))  float    v8f;

#define NUM_EMB   1024
#define DIM       64
#define N_ROWS    (32 * 4096)
#define N_TILES   (NUM_EMB / 16)   // 64 codebook tiles of 16 columns

// ---- workspace layout (bytes) ----
// 0     : float loss_accum
// 64    : int   counts[1024]
// 8192  : _Float16 bfrag[64 tiles][2 kchunks][32 lanes][16 halves]  = 128 KB
#define WS_COUNTS_OFF 64
#define WS_BFRAG_OFF  8192

// ---------------------------------------------------------------------------
// Kernel 0: zero the accumulators we rely on (ws is poisoned with 0xAA).
// ---------------------------------------------------------------------------
__global__ __launch_bounds__(256) void vq_init_kernel(float* loss, int* counts) {
    int t = threadIdx.x;
    if (t == 0) loss[0] = 0.0f;
    for (int i = t; i < NUM_EMB; i += 256) counts[i] = 0;
}

// ---------------------------------------------------------------------------
// Kernel 1: L2-normalize embeddings -> f16, pre-swizzled into the WMMA
// B-fragment layout: bfrag[((t*2 + f)*32 + lane)*16 + i]
//   lane = sel*16 + (n & 15),  i<8 : K = f*32 + sel*8 + i
//                              i>=8: K = f*32 + 16 + sel*8 + (i-8)
// ---------------------------------------------------------------------------
__global__ __launch_bounds__(256) void vq_prep_kernel(const float* __restrict__ emb,
                                                      _Float16* __restrict__ bfrag) {
    int r = blockIdx.x * 256 + threadIdx.x;
    if (r >= NUM_EMB) return;
    const float* e = emb + (size_t)r * DIM;

    float v[DIM];
    float ss = 0.0f;
#pragma unroll
    for (int i = 0; i < DIM; ++i) { v[i] = e[i]; ss += v[i] * v[i]; }
    float rn = 1.0f / fmaxf(sqrtf(ss), 1e-12f);

    int t  = r >> 4;
    int lo = r & 15;
#pragma unroll
    for (int f = 0; f < 2; ++f) {
#pragma unroll
        for (int sel = 0; sel < 2; ++sel) {
            _Float16* dst = bfrag + (((size_t)(t * 2 + f) * 32) + sel * 16 + lo) * 16;
#pragma unroll
            for (int i = 0; i < 16; ++i) {
                int K = f * 32 + ((i < 8) ? (sel * 8 + i) : (16 + sel * 8 + (i - 8)));
                dst[i] = (_Float16)(v[K] * rn);
            }
        }
    }
}

// ---------------------------------------------------------------------------
// Kernel 2 (fused): WMMA argmax + gather + loss + counts.
// 256 threads = 8 waves; each wave handles TWO 16-row M-tiles (32 rows).
// Full 128 KB pre-swizzled B-fragment array staged in LDS per workgroup;
// A-fragment build overlaps the staging; B reads are software-pipelined one
// tile ahead so ds_load latency hides under the WMMAs + argmax chain.
// ---------------------------------------------------------------------------
__global__ __launch_bounds__(256) void vq_argmax_fused_kernel(const float* __restrict__ flat,
                                                              const float* __restrict__ emb,
                                                              const _Float16* __restrict__ bfrag,
                                                              float* __restrict__ idx_out,
                                                              float* __restrict__ qout,
                                                              float* __restrict__ loss,
                                                              int* __restrict__ counts) {
    __shared__ v16h lds_b[N_TILES * 2 * 32];   // 4096 * 32 B = 128 KB

    // Issue cooperative stage: global -> LDS (16 x 32B per thread).
    const v16h* bsrc = (const v16h*)bfrag;
    for (int i = threadIdx.x; i < N_TILES * 2 * 32; i += 256) {
        lds_b[i] = bsrc[i];
    }

    const int lane = threadIdx.x & 31;
    const int wave = threadIdx.x >> 5;
    const int base = blockIdx.x * 256 + wave * 32;   // first of 32 rows for this wave
    const int m    = lane & 15;                      // row within a 16-row tile
    const int sel  = lane >> 4;                      // half selector (K pattern)

    // Build the two A fragments (K=0..31 and K=32..63 patterns) for a row.
    // Pure global-memory work: overlaps the LDS staging above (barrier below).
    auto build_a = [&](int row, v16h& A0, v16h& A1) {
        const float* rowp = flat + (size_t)row * DIM;
        float xa[32];
#pragma unroll
        for (int g = 0; g < 4; ++g) {
            float4 a = *(const float4*)(rowp + g * 16 + sel * 8 + 0);
            float4 b = *(const float4*)(rowp + g * 16 + sel * 8 + 4);
            xa[g * 8 + 0] = a.x; xa[g * 8 + 1] = a.y; xa[g * 8 + 2] = a.z; xa[g * 8 + 3] = a.w;
            xa[g * 8 + 4] = b.x; xa[g * 8 + 5] = b.y; xa[g * 8 + 6] = b.z; xa[g * 8 + 7] = b.w;
        }
        float ss = 0.0f;
#pragma unroll
        for (int i = 0; i < 32; ++i) ss += xa[i] * xa[i];
        ss += __shfl_xor(ss, 16, 32);                // other half has the other 32 elems
        float rn = 1.0f / fmaxf(sqrtf(ss), 1e-12f);
#pragma unroll
        for (int i = 0; i < 16; ++i) {
            A0[i] = (_Float16)(xa[i] * rn);
            A1[i] = (_Float16)(xa[16 + i] * rn);
        }
    };

    v16h a00, a01, a10, a11;
    build_a(base + m,      a00, a01);                // M-tile 0
    build_a(base + 16 + m, a10, a11);                // M-tile 1

    __syncthreads();                                 // LDS stage complete

    float bestV0[8], bestV1[8];
    int   bestI0[8], bestI1[8];
#pragma unroll
    for (int r = 0; r < 8; ++r) {
        bestV0[r] = -3.0e38f; bestI0[r] = 0;
        bestV1[r] = -3.0e38f; bestI1[r] = 0;
    }

    // Software-pipelined tile loop: prefetch tile t+1's B fragments (wrapped
    // index -> always-valid LDS address, no branch) while computing tile t.
    v16h b0 = lds_b[0 * 32 + lane];
    v16h b1 = lds_b[1 * 32 + lane];
    for (int t = 0; t < N_TILES; ++t) {
        const int tn = (t + 1) & (N_TILES - 1);
        v16h nb0 = lds_b[(tn * 2 + 0) * 32 + lane];  // ds_load_b128 x2
        v16h nb1 = lds_b[(tn * 2 + 1) * 32 + lane];

        v8f c0 = {};
        v8f c1 = {};
        c0 = __builtin_amdgcn_wmma_f32_16x16x32_f16(false, a00, false, b0, (short)0, c0, false, false);
        c1 = __builtin_amdgcn_wmma_f32_16x16x32_f16(false, a10, false, b0, (short)0, c1, false, false);
        c0 = __builtin_amdgcn_wmma_f32_16x16x32_f16(false, a01, false, b1, (short)0, c0, false, false);
        c1 = __builtin_amdgcn_wmma_f32_16x16x32_f16(false, a11, false, b1, (short)0, c1, false, false);

        const int cand = t * 16 + m;                 // codebook column this lane sees
#pragma unroll
        for (int r = 0; r < 8; ++r) {
            float v0 = c0[r];
            if (v0 > bestV0[r]) { bestV0[r] = v0; bestI0[r] = cand; }
            float v1 = c1[r];
            if (v1 > bestV1[r]) { bestV1[r] = v1; bestI1[r] = cand; }
        }
        b0 = nb0;
        b1 = nb1;
    }

    // Butterfly max-reduce across each 16-lane half (rows 0-7 of a tile live
    // in lanes 0-15, rows 8-15 in lanes 16-31 per the C-matrix layout).
#pragma unroll
    for (int msk = 8; msk >= 1; msk >>= 1) {
#pragma unroll
        for (int r = 0; r < 8; ++r) {
            float ov0 = __shfl_xor(bestV0[r], msk, 32);
            int   oi0 = __shfl_xor(bestI0[r], msk, 32);
            if (ov0 > bestV0[r] || (ov0 == bestV0[r] && oi0 < bestI0[r])) {
                bestV0[r] = ov0; bestI0[r] = oi0;
            }
            float ov1 = __shfl_xor(bestV1[r], msk, 32);
            int   oi1 = __shfl_xor(bestI1[r], msk, 32);
            if (ov1 > bestV1[r] || (ov1 == bestV1[r] && oi1 < bestI1[r])) {
                bestV1[r] = ov1; bestI1[r] = oi1;
            }
        }
    }

    // Route each row's winning index back to the two lanes that own the row.
    // Constant-index select tree (r = m & 7), then cross-half exchange.
    auto pick = [&](const int* arr) -> int {
        int a0 = (m & 1) ? arr[1] : arr[0];
        int a1 = (m & 1) ? arr[3] : arr[2];
        int a2 = (m & 1) ? arr[5] : arr[4];
        int a3 = (m & 1) ? arr[7] : arr[6];
        int b0s = (m & 2) ? a1 : a0;
        int b1s = (m & 2) ? a3 : a2;
        return (m & 4) ? b1s : b0s;
    };
    int loc0 = pick(bestI0);
    int loc1 = pick(bestI1);
    int oth0 = __shfl_xor(loc0, 16, 32);
    int oth1 = __shfl_xor(loc1, 16, 32);
    const bool own = ((m >> 3) == sel);              // row's result lives in my half?
    const int  k0  = own ? loc0 : oth0;              // winner for row base+m
    const int  k1  = own ? loc1 : oth1;              // winner for row base+16+m

    if (sel == 0) {                                  // one writer per row
        idx_out[base + m]      = (float)k0;
        idx_out[base + 16 + m] = (float)k1;
        atomicAdd(&counts[k0], 1);
        atomicAdd(&counts[k1], 1);
    }

    // Gather raw embedding rows, write quantized output (each lane covers its
    // K-pattern slice: 4 groups of 8 floats), accumulate squared error.
    float s = 0.0f;
#pragma unroll
    for (int tt = 0; tt < 2; ++tt) {
        const int    k   = tt ? k1 : k0;
        const size_t row = (size_t)(base + tt * 16 + m);
        const float* ep  = emb  + (size_t)k * DIM;
        const float* xp  = flat + row * DIM;
        float*       qp  = qout + row * DIM;
#pragma unroll
        for (int g = 0; g < 4; ++g) {
            const int off = g * 16 + sel * 8;
            float4 e0 = *(const float4*)(ep + off);
            float4 e1 = *(const float4*)(ep + off + 4);
            float4 x0 = *(const float4*)(xp + off);
            float4 x1 = *(const float4*)(xp + off + 4);
            *(float4*)(qp + off)     = e0;           // STE: values == quantized
            *(float4*)(qp + off + 4) = e1;
            float d;
            d = e0.x - x0.x; s += d * d;
            d = e0.y - x0.y; s += d * d;
            d = e0.z - x0.z; s += d * d;
            d = e0.w - x0.w; s += d * d;
            d = e1.x - x1.x; s += d * d;
            d = e1.y - x1.y; s += d * d;
            d = e1.z - x1.z; s += d * d;
            d = e1.w - x1.w; s += d * d;
        }
    }
#pragma unroll
    for (int msk = 16; msk >= 1; msk >>= 1) s += __shfl_xor(s, msk, 32);
    if (lane == 0) atomicAdd(loss, s);
}

// ---------------------------------------------------------------------------
// Kernel 3: finalize scalar stats.
// ---------------------------------------------------------------------------
__global__ __launch_bounds__(256) void vq_final_kernel(const int* __restrict__ counts,
                                                       const float* __restrict__ loss,
                                                       float* __restrict__ out3) {
    __shared__ float s_ent[8];
    __shared__ float s_use[8];
    const int t = threadIdx.x;
    float ent = 0.0f, used = 0.0f;
    for (int i = t; i < NUM_EMB; i += 256) {
        float p = (float)counts[i] / (float)N_ROWS;
        ent  += p * logf(p + 1e-10f);
        used += (p > 0.0f) ? 1.0f : 0.0f;
    }
#pragma unroll
    for (int msk = 16; msk >= 1; msk >>= 1) {
        ent  += __shfl_xor(ent,  msk, 32);
        used += __shfl_xor(used, msk, 32);
    }
    if ((t & 31) == 0) { s_ent[t >> 5] = ent; s_use[t >> 5] = used; }
    __syncthreads();
    if (t == 0) {
        float E = 0.0f, U = 0.0f;
        for (int i = 0; i < 8; ++i) { E += s_ent[i]; U += s_use[i]; }
        out3[0] = 1.25f * loss[0] / (float)((size_t)N_ROWS * DIM);  // vq_loss
        out3[1] = expf(-E);                                         // perplexity
        out3[2] = U / (float)NUM_EMB;                               // codebook usage
    }
}

// ---------------------------------------------------------------------------
extern "C" void kernel_launch(void* const* d_in, const int* in_sizes, int n_in,
                              void* d_out, int out_size, void* d_ws, size_t ws_size,
                              hipStream_t stream) {
    const float* flat = (const float*)d_in[0];   // [32,4096,64]
    const float* emb  = (const float*)d_in[1];   // [1024,64]

    float* out      = (float*)d_out;
    float* qout     = out;                              // 8388608 floats
    float* idx_out  = out + (size_t)N_ROWS * DIM;       // 131072 floats
    float* out3     = idx_out + N_ROWS;                 // 3 floats

    float*    loss   = (float*)d_ws;
    int*      counts = (int*)((char*)d_ws + WS_COUNTS_OFF);
    _Float16* bfrag  = (_Float16*)((char*)d_ws + WS_BFRAG_OFF);

    vq_init_kernel<<<1, 256, 0, stream>>>(loss, counts);
    vq_prep_kernel<<<NUM_EMB / 256, 256, 0, stream>>>(emb, bfrag);
    vq_argmax_fused_kernel<<<N_ROWS / 256, 256, 0, stream>>>(flat, emb, bfrag,
                                                             idx_out, qout, loss, counts);
    vq_final_kernel<<<1, 256, 0, stream>>>(counts, loss, out3);
}
